// GRUTimeSeries_27608049778811
// MI455X (gfx1250) — matrix operand order repaired and checked by reading, not compiled, not run on hardware
//
#include <hip/hip_runtime.h>
#include <hip/hip_bf16.h>

// ---------------------------------------------------------------------------
// GRU (2 layers, B=256, T=512, H=512) + 3-layer MLP head, for MI455X (gfx1250)
// - All GEMMs: v_wmma_f32_16x16x32_bf16 (wave32, 16x16 tile/wave, f32 accum)
// - Layer-0 step t and layer-1 step t-1 fused in ONE kernel launch (they read
//   the same h0 ping-pong buffer), halving scan launches to 513.
// - A-operand tiles staged into LDS via the Tensor Data Mover
//   (tensor_load_to_lds + s_wait_tensorcnt), shared by all 8 waves per block.
// ---------------------------------------------------------------------------

typedef __attribute__((ext_vector_type(16))) __bf16 v16bf;
typedef __attribute__((ext_vector_type(8)))  float  v8f;
typedef __attribute__((ext_vector_type(4)))  unsigned int v4u32;
typedef __attribute__((ext_vector_type(8)))  int    v8i32;
typedef __attribute__((ext_vector_type(4)))  int    v4i32;

#define GRU_B   256
#define GRU_T   512
#define GRU_H   512
#define GRU_G   1536   // 3*H
#define GRU_OUT 256

union UFrag {
    v16bf v;
    uint4 q[2];
};

__device__ __forceinline__ unsigned short f32_to_bf16_rne(float f) {
    unsigned int u = __float_as_uint(f);
    unsigned int lsb = (u >> 16) & 1u;
    u += 0x7fffu + lsb;
    return (unsigned short)(u >> 16);
}

__device__ __forceinline__ float sigmoid_f(float x) {
    return 1.0f / (1.0f + __expf(-x));
}

// A fragment (16x32 bf16) from row-major A, per CDNA5 ISA layout:
// lane<16 -> row=lane, K=[0..7],[16..23]; lane>=16 -> row=lane-16, K=[8..15],[24..31]
__device__ __forceinline__ v16bf load_a_frag(const unsigned short* A, int lda,
                                             int row0, int k0, int lane) {
    const int laneHi = lane >> 4;
    const int r      = row0 + (lane & 15);
    const unsigned short* p = A + (size_t)r * lda + k0 + laneHi * 8;
    UFrag f;
    f.q[0] = *(const uint4*)(p);
    f.q[1] = *(const uint4*)(p + 16);
    return f.v;
}

// A fragment from an LDS-staged 16x512 tile (row-major, row stride 512)
__device__ __forceinline__ v16bf load_a_frag_lds(const unsigned short* L,
                                                 int k0, int lane) {
    const int laneHi = lane >> 4;
    const int r      = lane & 15;
    const unsigned short* p = L + r * GRU_H + k0 + laneHi * 8;
    UFrag f;
    f.q[0] = *(const uint4*)(p);        // -> ds_load_b128
    f.q[1] = *(const uint4*)(p + 16);
    return f.v;
}

// B fragment (32x16) = W^T tile from row-major W (N,K):
// lane<16 -> col=lane, K=[0..7],[16..23]; lane>=16 -> col=lane-16, K=[8..15],[24..31]
__device__ __forceinline__ v16bf load_b_frag(const unsigned short* W, int ldw,
                                             int ncol0, int k0, int lane) {
    const int laneHi = lane >> 4;
    const int rw     = ncol0 + (lane & 15);
    const unsigned short* p = W + (size_t)rw * ldw + k0 + laneHi * 8;
    UFrag f;
    f.q[0] = *(const uint4*)(p);
    f.q[1] = *(const uint4*)(p + 16);
    return f.v;
}

__device__ __forceinline__ v8f wmma_bf16(v16bf a, v16bf b, v8f c) {
    return __builtin_amdgcn_wmma_f32_16x16x32_bf16(
        false, a, false, b, (short)0, c, false, false);
}

// ---------------------------------------------------------------------------
// TDM: DMA a 2-D bf16 tile (rows x 512, row stride 512 elems) into LDS.
// D# packing per CDNA5 ISA.
// ---------------------------------------------------------------------------
__device__ __forceinline__ void tdm_load_tile_2d(unsigned int lds_addr,
                                                 const unsigned short* gptr,
                                                 int rows) {
    unsigned long long ga = (unsigned long long)(uintptr_t)gptr;
    v4u32 g0;
    g0[0] = 1u;                                   // count=1, user mode
    g0[1] = lds_addr;                             // LDS byte address
    g0[2] = (unsigned int)(ga & 0xFFFFFFFFull);   // global_addr[31:0]
    g0[3] = (unsigned int)((ga >> 32) & 0x1FFFFFFull) | (2u << 30); // type=2
    v8i32 g1;
    g1[0] = 0x00010000;                   // workgroup_mask=0, data_size=1 (2B)
    g1[1] = (int)(512u << 16);            // tensor_dim0[15:0] @ bits[63:48]
    g1[2] = (int)((unsigned)rows << 16);  // tensor_dim1[15:0] @ bits[95:80]
    g1[3] = (int)(512u << 16);            // tile_dim0=512 @ bits[127:112]
    g1[4] = rows;                         // tile_dim1 @ bits[143:128]
    g1[5] = 512;                          // tensor_dim0_stride[31:0] @ [191:160]
    g1[6] = 0;
    g1[7] = 0;
    v4i32 gz = {0, 0, 0, 0};
#if defined(__clang_major__) && (__clang_major__ >= 23)
    v8i32 gz8 = {0, 0, 0, 0, 0, 0, 0, 0};
    __builtin_amdgcn_tensor_load_to_lds(g0, g1, gz, gz, gz8, 0);
#else
    __builtin_amdgcn_tensor_load_to_lds(g0, g1, gz, gz, 0);
#endif
}

// ---------------------------------------------------------------------------
// f32 -> bf16 conversion (grid-stride)
// ---------------------------------------------------------------------------
__global__ void cvt_f32_to_bf16_kernel(const float* __restrict__ src,
                                       unsigned short* __restrict__ dst, int n) {
    int i = blockIdx.x * blockDim.x + threadIdx.x;
    int stride = gridDim.x * blockDim.x;
    for (; i < n; i += stride) dst[i] = f32_to_bf16_rne(src[i]);
}

// ---------------------------------------------------------------------------
// Fused GRU step: at launch t, waves 0-3 run layer-0 step t (if t<T) and
// waves 4-7 run layer-1 step t-1 (if t>=1). Both layers read h0b_rd, so one
// TDM-staged LDS tile serves all 8 waves. Grid: 128 blocks x 256 threads.
//   block -> mt = blockIdx>>3 (batch rows), nt0 = (blockIdx&7)*4 (col tiles)
// ---------------------------------------------------------------------------
__global__ void gru_fused_step_kernel(int t,
        const unsigned short* __restrict__ h0b_rd,   // h0 state t   (bf16)
        const float*          __restrict__ h0f_rd,   // h0 state t   (f32)
        unsigned short*       __restrict__ h0b_wr,   // h0 state t+1
        float*                __restrict__ h0f_wr,
        const unsigned short* __restrict__ h1b_rd,   // h1 state t-1 (bf16)
        const float*          __restrict__ h1f_rd,
        unsigned short*       __restrict__ h1b_wr,   // h1 state t
        float*                __restrict__ h1f_wr,
        const unsigned short* __restrict__ Whh0b,    // (1536,512) bf16
        const unsigned short* __restrict__ Wih1b,
        const unsigned short* __restrict__ Whh1b,
        const float* __restrict__ x,                 // (B,T,1)
        const float* __restrict__ Wih0,              // (1536,1)
        const float* __restrict__ bih0, const float* __restrict__ bhh0,
        const float* __restrict__ bih1, const float* __restrict__ bhh1)
{
    __shared__ __align__(16) unsigned short lds0[16 * GRU_H]; // h0 A-tile
    __shared__ __align__(16) unsigned short lds1[16 * GRU_H]; // h1 A-tile

    const int lane = threadIdx.x & 31;
    const int wave = threadIdx.x >> 5;            // 0..7
    const int mt   = blockIdx.x >> 3;             // 0..15
    const int nt   = ((blockIdx.x & 7) << 2) + (wave & 3);  // 0..31
    const int row0 = mt * 16;
    const int col0 = nt * 16;

    // --- stage both A tiles via Tensor Data Mover (wave 0 only) ---
    if (wave == 0) {
        tdm_load_tile_2d((unsigned int)(uintptr_t)&lds0[0],
                         h0b_rd + (size_t)row0 * GRU_H, 16);
        tdm_load_tile_2d((unsigned int)(uintptr_t)&lds1[0],
                         h1b_rd + (size_t)row0 * GRU_H, 16);
        __builtin_amdgcn_s_wait_tensorcnt(0);
    }
    __syncthreads();

    const int laneHi = lane >> 4;
    const int col = col0 + (lane & 15);

    if (wave < 4) {
        // ---------------- layer 0, step t ----------------
        if (t < GRU_T) {
            v8f accR = {}, accZ = {}, accN = {};
            for (int k0 = 0; k0 < GRU_H; k0 += 32) {
                if (k0 + 32 < GRU_H)
                    __builtin_prefetch(
                        Whh0b + (size_t)(col0 + (lane & 15)) * GRU_H + k0 + 32, 0, 1);
                v16bf a  = load_a_frag_lds(lds0, k0, lane);
                v16bf bR = load_b_frag(Whh0b, GRU_H, 0 * GRU_H + col0, k0, lane);
                accR = wmma_bf16(a, bR, accR);
                v16bf bZ = load_b_frag(Whh0b, GRU_H, 1 * GRU_H + col0, k0, lane);
                accZ = wmma_bf16(a, bZ, accZ);
                v16bf bN = load_b_frag(Whh0b, GRU_H, 2 * GRU_H + col0, k0, lane);
                accN = wmma_bf16(a, bN, accN);
            }
            const float wxr = Wih0[col],           bir  = bih0[col],           bhr = bhh0[col];
            const float wxz = Wih0[GRU_H + col],   biz  = bih0[GRU_H + col],   bhz = bhh0[GRU_H + col];
            const float wxn = Wih0[2*GRU_H + col], bin_ = bih0[2*GRU_H + col], bhn = bhh0[2*GRU_H + col];
#pragma unroll
            for (int i = 0; i < 8; ++i) {
                const int b = row0 + i + laneHi * 8;
                const float xv = x[(size_t)b * GRU_T + t];
                const float r  = sigmoid_f((xv * wxr + bir) + (accR[i] + bhr));
                const float z  = sigmoid_f((xv * wxz + biz) + (accZ[i] + bhz));
                const float nn = tanhf((xv * wxn + bin_) + r * (accN[i] + bhn));
                const float hp = h0f_rd[(size_t)b * GRU_H + col];
                const float hn = (1.0f - z) * nn + z * hp;
                h0f_wr[(size_t)b * GRU_H + col] = hn;
                h0b_wr[(size_t)b * GRU_H + col] = f32_to_bf16_rne(hn);
            }
        }
    } else {
        // ---------------- layer 1, step t-1 ----------------
        if (t >= 1) {
            v8f axR = {}, axZ = {}, axN = {};
            v8f ahR = {}, ahZ = {}, ahN = {};
            for (int k0 = 0; k0 < GRU_H; k0 += 32) {
                if (k0 + 32 < GRU_H)
                    __builtin_prefetch(
                        Wih1b + (size_t)(col0 + (lane & 15)) * GRU_H + k0 + 32, 0, 1);
                {   // input projection: A = hs0[t-1] (= h0 state t), B = Wih1
                    v16bf a  = load_a_frag_lds(lds0, k0, lane);
                    v16bf bR = load_b_frag(Wih1b, GRU_H, 0 * GRU_H + col0, k0, lane);
                    axR = wmma_bf16(a, bR, axR);
                    v16bf bZ = load_b_frag(Wih1b, GRU_H, 1 * GRU_H + col0, k0, lane);
                    axZ = wmma_bf16(a, bZ, axZ);
                    v16bf bN = load_b_frag(Wih1b, GRU_H, 2 * GRU_H + col0, k0, lane);
                    axN = wmma_bf16(a, bN, axN);
                }
                {   // recurrent projection: A = h1 state t-1, B = Whh1
                    v16bf a  = load_a_frag_lds(lds1, k0, lane);
                    v16bf bR = load_b_frag(Whh1b, GRU_H, 0 * GRU_H + col0, k0, lane);
                    ahR = wmma_bf16(a, bR, ahR);
                    v16bf bZ = load_b_frag(Whh1b, GRU_H, 1 * GRU_H + col0, k0, lane);
                    ahZ = wmma_bf16(a, bZ, ahZ);
                    v16bf bN = load_b_frag(Whh1b, GRU_H, 2 * GRU_H + col0, k0, lane);
                    ahN = wmma_bf16(a, bN, ahN);
                }
            }
            const float bir  = bih1[col],           bhr = bhh1[col];
            const float biz  = bih1[GRU_H + col],   bhz = bhh1[GRU_H + col];
            const float bin_ = bih1[2*GRU_H + col], bhn = bhh1[2*GRU_H + col];
#pragma unroll
            for (int i = 0; i < 8; ++i) {
                const int b = row0 + i + laneHi * 8;
                const float r  = sigmoid_f((axR[i] + bir) + (ahR[i] + bhr));
                const float z  = sigmoid_f((axZ[i] + biz) + (ahZ[i] + bhz));
                const float nn = tanhf((axN[i] + bin_) + r * (ahN[i] + bhn));
                const float hp = h1f_rd[(size_t)b * GRU_H + col];
                const float hn = (1.0f - z) * nn + z * hp;
                h1f_wr[(size_t)b * GRU_H + col] = hn;
                h1b_wr[(size_t)b * GRU_H + col] = f32_to_bf16_rne(hn);
            }
        }
    }
}

// ---------------------------------------------------------------------------
// Dense layer: C(256,N) = act(A_bf16(256,512) @ W_bf16(N,512)^T + bias)
// act==0: ReLU -> bf16 out.  act==1: sigmoid -> f32 d_out.
// ---------------------------------------------------------------------------
__global__ void dense_wmma_kernel(const unsigned short* __restrict__ Ab,
                                  const unsigned short* __restrict__ Wb,
                                  const float* __restrict__ bias,
                                  int N, int act,
                                  unsigned short* __restrict__ outb,
                                  float* __restrict__ outf)
{
    const int lane = threadIdx.x & 31;
    const int wave = threadIdx.x >> 5;
    const int tile = blockIdx.x * (blockDim.x >> 5) + wave;
    const int ntiles = N >> 4;
    const int mt = tile / ntiles;
    const int nt = tile % ntiles;
    const int row0 = mt * 16, col0 = nt * 16;

    v8f acc = {};
    for (int k0 = 0; k0 < GRU_H; k0 += 32) {
        if (k0 + 32 < GRU_H)
            __builtin_prefetch(Ab + (size_t)row0 * GRU_H + k0 + 32, 0, 1);
        v16bf a = load_a_frag(Ab, GRU_H, row0, k0, lane);
        v16bf b = load_b_frag(Wb, GRU_H, col0, k0, lane);
        acc = wmma_bf16(a, b, acc);
    }

    const int laneHi = lane >> 4;
    const int col = col0 + (lane & 15);
    const float bs = bias[col];
#pragma unroll
    for (int i = 0; i < 8; ++i) {
        const int brow = row0 + i + laneHi * 8;
        float v = acc[i] + bs;
        if (act == 0) {
            v = fmaxf(v, 0.0f);
            outb[(size_t)brow * N + col] = f32_to_bf16_rne(v);
        } else {
            outf[(size_t)brow * N + col] = sigmoid_f(v);
        }
    }
}

// ---------------------------------------------------------------------------
// Host side
// ---------------------------------------------------------------------------
extern "C" void kernel_launch(void* const* d_in, const int* in_sizes, int n_in,
                              void* d_out, int out_size, void* d_ws, size_t ws_size,
                              hipStream_t stream) {
    (void)in_sizes; (void)n_in; (void)out_size; (void)ws_size;

    const float* x     = (const float*)d_in[0];
    const float* Wih0  = (const float*)d_in[1];
    const float* Whh0  = (const float*)d_in[2];
    const float* bih0  = (const float*)d_in[3];
    const float* bhh0  = (const float*)d_in[4];
    const float* Wih1  = (const float*)d_in[5];
    const float* Whh1  = (const float*)d_in[6];
    const float* bih1  = (const float*)d_in[7];
    const float* bhh1  = (const float*)d_in[8];
    const float* W1    = (const float*)d_in[9];
    const float* b1    = (const float*)d_in[10];
    const float* W2    = (const float*)d_in[11];
    const float* b2    = (const float*)d_in[12];
    const float* W3    = (const float*)d_in[13];
    const float* b3    = (const float*)d_in[14];

    char* base = (char*)d_ws;
    size_t off = 0;
    auto alloc = [&](size_t bytes) -> char* {
        off = (off + 255) & ~(size_t)255;
        char* p = base + off;
        off += bytes;
        return p;
    };
    const size_t nGH = (size_t)GRU_G * GRU_H;
    const size_t nHH = (size_t)GRU_H * GRU_H;
    const size_t nOH = (size_t)GRU_OUT * GRU_H;
    const size_t nBH = (size_t)GRU_B * GRU_H;

    unsigned short* Whh0b = (unsigned short*)alloc(nGH * 2);
    unsigned short* Wih1b = (unsigned short*)alloc(nGH * 2);
    unsigned short* Whh1b = (unsigned short*)alloc(nGH * 2);
    unsigned short* W1b   = (unsigned short*)alloc(nHH * 2);
    unsigned short* W2b   = (unsigned short*)alloc(nHH * 2);
    unsigned short* W3b   = (unsigned short*)alloc(nOH * 2);

    float*          h0f[2] = { (float*)alloc(nBH * 4), (float*)alloc(nBH * 4) };
    unsigned short* h0b[2] = { (unsigned short*)alloc(nBH * 2), (unsigned short*)alloc(nBH * 2) };
    float*          h1f[2] = { (float*)alloc(nBH * 4), (float*)alloc(nBH * 4) };
    unsigned short* h1b[2] = { (unsigned short*)alloc(nBH * 2), (unsigned short*)alloc(nBH * 2) };
    unsigned short* y1b = (unsigned short*)alloc(nBH * 2);
    unsigned short* y2b = (unsigned short*)alloc(nBH * 2);

    // weight conversion f32 -> bf16
    cvt_f32_to_bf16_kernel<<<512, 256, 0, stream>>>(Whh0, Whh0b, (int)nGH);
    cvt_f32_to_bf16_kernel<<<512, 256, 0, stream>>>(Wih1, Wih1b, (int)nGH);
    cvt_f32_to_bf16_kernel<<<512, 256, 0, stream>>>(Whh1, Whh1b, (int)nGH);
    cvt_f32_to_bf16_kernel<<<256, 256, 0, stream>>>(W1, W1b, (int)nHH);
    cvt_f32_to_bf16_kernel<<<256, 256, 0, stream>>>(W2, W2b, (int)nHH);
    cvt_f32_to_bf16_kernel<<<128, 256, 0, stream>>>(W3, W3b, (int)nOH);

    // zero-init hidden state (both ping-pong buffers, f32 + bf16)
    hipMemsetAsync(h0f[0], 0, nBH * 4 * 2, stream);
    hipMemsetAsync(h0b[0], 0, nBH * 2 * 2, stream);
    hipMemsetAsync(h1f[0], 0, nBH * 4 * 2, stream);
    hipMemsetAsync(h1b[0], 0, nBH * 2 * 2, stream);

    // fused pipelined scan: launch t runs L0 step t and L1 step t-1.
    // h0 state s lives in h0*[s&1]; h1 state s is written to h1*[s&1].
    for (int t = 0; t <= GRU_T; ++t) {
        gru_fused_step_kernel<<<128, 256, 0, stream>>>(
            t,
            h0b[t & 1],       h0f[t & 1],        // h0 state t (read)
            h0b[(t + 1) & 1], h0f[(t + 1) & 1],  // h0 state t+1 (write)
            h1b[(t + 1) & 1], h1f[(t + 1) & 1],  // h1 state t-1 (read)
            h1b[t & 1],       h1f[t & 1],        // h1 state t   (write)
            Whh0b, Wih1b, Whh1b,
            x, Wih0, bih0, bhh0, bih1, bhh1);
    }
    // final h1 state (after step 511) was written at t=512 into index 0.

    // MLP head
    dense_wmma_kernel<<<128, 128, 0, stream>>>(h1b[0], W1b, b1, GRU_H, 0, y1b, nullptr);
    dense_wmma_kernel<<<128, 128, 0, stream>>>(y1b,    W2b, b2, GRU_H, 0, y2b, nullptr);
    dense_wmma_kernel<<<64, 128, 0, stream>>>(y2b,     W3b, b3, GRU_OUT, 1, nullptr, (float*)d_out);
}
